// GraphSAGEConv_2319282339967
// MI455X (gfx1250) — matrix operand order repaired and checked by reading, not compiled, not run on hardware
//
#include <hip/hip_runtime.h>
#include <hip/hip_bf16.h>

// ---------------------------------------------------------------------------
// GraphSAGE conv for MI455X (gfx1250, wave32, WMMA, TDM).
//   scatter_mean (float atomics, L2-resident) ->
//   hcat bf16 [N,256] ->
//   GEMM1 bf16 WMMA + bias + relu -> h bf16 [N,512] ->
//   GEMM2 bf16 WMMA + bias -> out f32 [N,128]
// A-tiles are staged into LDS by the Tensor Data Mover (TENSOR_LOAD_TO_LDS,
// TENSORcnt-tracked) when the builtin is available; vector-copy fallback
// otherwise.
// ---------------------------------------------------------------------------

typedef __bf16 bf16_t;
typedef __attribute__((ext_vector_type(16))) __bf16 v16bf;
typedef __attribute__((ext_vector_type(8)))  __bf16 v8bf;
typedef __attribute__((ext_vector_type(8)))  float  v8f;

typedef __attribute__((ext_vector_type(4))) unsigned int u32x4;
typedef __attribute__((ext_vector_type(8))) int          i32x8;
typedef __attribute__((ext_vector_type(4))) int          i32x4;

union ABfrag { v16bf v; v8bf h[2]; };

static constexpr int C     = 128;   // in channels
static constexpr int TWO_C = 256;   // concat width (K of GEMM1)
static constexpr int HDIM  = 512;   // hidden (N of GEMM1, K of GEMM2)
static constexpr int ODIM  = 128;   // out channels (N of GEMM2)

#define SAGE_HAVE_TDM __has_builtin(__builtin_amdgcn_tensor_load_to_lds)

#if SAGE_HAVE_TDM
// ---------------------------------------------------------------------------
// 1-D TDM load: copy n_elems8 * 8 bytes from gaddr into LDS at lds_addr.
// Descriptor bit layout per CDNA5 ISA section 8.3/8.4:
//   group0: [1:0]=count=1, [63:32]=lds_addr, [120:64]=global_addr,
//           [127:126]=type=2
//   group1: [17:16]=data_size (3 -> 8B), [79:48]=tensor_dim0,
//           [127:112]=tile_dim0; rest 0 (1-D tile)
//   groups 2/3: zero (<=2-D tensor)
// Issue from ONE wave per block (TDM ignores EXEC; one instruction = one DMA).
// ---------------------------------------------------------------------------
__device__ inline void sage_tdm_load_1d(unsigned int lds_addr,
                                        const void* gaddr,
                                        unsigned int n_elems8) {
    unsigned long long ga = (unsigned long long)gaddr;

    u32x4 g0;
    g0[0] = 1u;                                        // count=1, is_restore=0
    g0[1] = lds_addr;                                  // LDS byte address
    g0[2] = (unsigned int)ga;                          // global_addr[31:0]
    g0[3] = (unsigned int)((ga >> 32) & 0x01FFFFFFull) // global_addr[56:32]
          | (2u << 30);                                // type=2 ("image")

    union { unsigned long long q[4]; i32x8 v; } g1;
    g1.q[0] = (3ull << 16)                                   // data_size=8B
            | ((unsigned long long)(n_elems8 & 0xFFFFu) << 48); // tensor_dim0 lo
    g1.q[1] = (unsigned long long)(n_elems8 >> 16)           // tensor_dim0 hi
            | ((unsigned long long)(n_elems8 & 0xFFFFu) << 48); // tile_dim0
    g1.q[2] = 0ull;
    g1.q[3] = 0ull;

    i32x4 g2 = {0, 0, 0, 0};
    i32x4 g3 = {0, 0, 0, 0};

#if __clang_major__ >= 23
    i32x8 g4 = {0, 0, 0, 0, 0, 0, 0, 0};
    __builtin_amdgcn_tensor_load_to_lds(g0, g1.v, g2, g3, g4, 0);
#else
    __builtin_amdgcn_tensor_load_to_lds(g0, g1.v, g2, g3, 0);
#endif
}
#endif  // SAGE_HAVE_TDM

// Stage 16 contiguous rows (16*K bf16, contiguous in memory) into LDS.
// Tail-of-N handled by clamping the transfer; garbage rows never stored.
__device__ inline void sage_stage_tile(bf16_t* sA, const bf16_t* A,
                                       int row0, int K, int N) {
#if SAGE_HAVE_TDM
    unsigned long long total8 = ((unsigned long long)N * K) >> 2;  // bf16*2/8
    unsigned long long base8  = ((unsigned long long)row0 * K) >> 2;
    unsigned int want8 = (unsigned int)((16u * (unsigned)K) >> 2);
    unsigned int elems8 = (base8 + want8 <= total8)
                              ? want8
                              : (unsigned int)(total8 > base8 ? total8 - base8 : 0);
    if (threadIdx.x < 32) {                     // one issuing wave per block
        sage_tdm_load_1d((unsigned int)(unsigned long long)(uintptr_t)sA,
                         A + (size_t)row0 * K, elems8);
        __builtin_amdgcn_s_wait_tensorcnt(0);   // TENSORcnt == 0
    }
#else
    const uint4* g = (const uint4*)(A + (size_t)row0 * K);
    uint4* s = (uint4*)sA;
    const int nvec = 16 * K / 8;
    const long long limit = ((long long)N * K) >> 3;
    const long long base  = ((long long)row0 * K) >> 3;
    for (int i = threadIdx.x; i < nvec; i += blockDim.x) {
        uint4 z = make_uint4(0u, 0u, 0u, 0u);
        s[i] = (base + i < limit) ? g[i] : z;
    }
#endif
    __syncthreads();
}

// ---------------------------------------------------------------------------
// Edge scatter: 32 threads per edge, float4 per thread (4 channels).
// x and sum are ~5MB each -> stays in the 192MB L2; atomic-throughput bound.
// ---------------------------------------------------------------------------
__global__ void sage_scatter_kernel(const float* __restrict__ x,
                                    const long long* __restrict__ ei,
                                    float* __restrict__ sum,
                                    float* __restrict__ cnt,
                                    int E) {
    int gid = blockIdx.x * blockDim.x + threadIdx.x;
    int e = gid >> 5;
    if (e >= E) return;
    int q = gid & 31;
    int row = (int)ei[e];
    int col = (int)ei[(long long)E + e];
    const float4 v = ((const float4*)(x + (size_t)col * C))[q];
    float* d = sum + (size_t)row * C + q * 4;
    atomicAdd(d + 0, v.x);
    atomicAdd(d + 1, v.y);
    atomicAdd(d + 2, v.z);
    atomicAdd(d + 3, v.w);
    if (q == 0) atomicAdd(cnt + row, 1.0f);
}

// ---------------------------------------------------------------------------
// Transpose + fp32->bf16 weight conversion: dst[n*K + k] = bf16(src[k*Ncols+n])
// ---------------------------------------------------------------------------
__global__ void sage_tconv_kernel(const float* __restrict__ src,
                                  bf16_t* __restrict__ dst,
                                  int K, int Ncols) {
    int gid = blockIdx.x * blockDim.x + threadIdx.x;
    if (gid >= K * Ncols) return;
    int n = gid / K;
    int k = gid - n * K;
    dst[gid] = (bf16_t)src[k * Ncols + n];
}

// ---------------------------------------------------------------------------
// Build concatenated [x | mean] row-major bf16 [N, 256]
// ---------------------------------------------------------------------------
__global__ void sage_hcat_kernel(const float* __restrict__ x,
                                 const float* __restrict__ sum,
                                 const float* __restrict__ cnt,
                                 bf16_t* __restrict__ hcat,
                                 int N) {
    int gid = blockIdx.x * blockDim.x + threadIdx.x;
    if (gid >= N * TWO_C) return;
    int n = gid >> 8;
    int c = gid & 255;
    float v;
    if (c < C) {
        v = x[(size_t)n * C + c];
    } else {
        float inv = 1.0f / fmaxf(cnt[n], 1.0f);
        v = sum[(size_t)n * C + (c - C)] * inv;
    }
    hcat[gid] = (bf16_t)v;
}

// ---------------------------------------------------------------------------
// GEMM1: h = relu(hcat @ w1 + b1).  A: [N,256] bf16, Bt: [512][256] bf16.
// Block = 256 thr = 8 waves; block tile = 16 rows x 128 cols; grid.y*128 cols.
// ---------------------------------------------------------------------------
__global__ void sage_gemm1_kernel(const bf16_t* __restrict__ A,
                                  const bf16_t* __restrict__ Bt,
                                  const float* __restrict__ bias,
                                  bf16_t* __restrict__ out,
                                  int N) {
    __shared__ bf16_t sA[16 * TWO_C];          // 8 KB
    const int row0 = blockIdx.x * 16;

    sage_stage_tile(sA, A, row0, TWO_C, N);    // TDM -> LDS (or copy fallback)

    const int lane = threadIdx.x & 31;
    const int wv   = threadIdx.x >> 5;
    const int hi   = lane >> 4;                 // lane half selects K sub-range
    const int m    = lane & 15;
    const int ncol = blockIdx.y * 128 + wv * 16 + m;

    v8f acc;
    const float bv = bias[ncol];
#pragma unroll
    for (int i = 0; i < 8; ++i) acc[i] = bv;

    const bf16_t* arow = sA + m * TWO_C + hi * 8;             // A frag (LDS)
    const bf16_t* brow = Bt + (size_t)ncol * TWO_C + hi * 8;  // B frag (global)

#pragma unroll
    for (int kt = 0; kt < TWO_C / 32; ++kt) {
        ABfrag a, b;
        // ISA 16-bit A 16x32 layout: lanes 0-15 K{0..7,16..23}, 16-31 K{8..15,24..31}
        a.h[0] = *(const v8bf*)(arow + kt * 32);
        a.h[1] = *(const v8bf*)(arow + kt * 32 + 16);
        // B 32x16: lanes 0-15 col=lane K{0..7,16..23}, lanes 16-31 col=lane-16 K{8..15,24..31}
        b.h[0] = *(const v8bf*)(brow + kt * 32);
        b.h[1] = *(const v8bf*)(brow + kt * 32 + 16);
        if (kt + 1 < TWO_C / 32)
            __builtin_prefetch(brow + (kt + 1) * 32, 0, 1);
        acc = __builtin_amdgcn_wmma_f32_16x16x32_bf16(
            false, a.v, false, b.v, (short)0, acc, false, false);
    }

    // C/D layout: VGPR v -> row M = v + 8*hi, col = lane&15 within tile
#pragma unroll
    for (int v = 0; v < 8; ++v) {
        int r = row0 + v + hi * 8;
        if (r < N) {
            float val = acc[v];
            val = val > 0.0f ? val : 0.0f;     // fused ReLU
            out[(size_t)r * HDIM + ncol] = (bf16_t)val;
        }
    }
}

// ---------------------------------------------------------------------------
// GEMM2: out = h @ w2 + b2.  A: [N,512] bf16, Bt: [128][512] bf16, out f32.
// Block tile = 16 rows x 128 cols (all of O); 16 K-steps of WMMA.
// ---------------------------------------------------------------------------
__global__ void sage_gemm2_kernel(const bf16_t* __restrict__ A,
                                  const bf16_t* __restrict__ Bt,
                                  const float* __restrict__ bias,
                                  float* __restrict__ out,
                                  int N) {
    __shared__ bf16_t sA[16 * HDIM];           // 16 KB
    const int row0 = blockIdx.x * 16;

    sage_stage_tile(sA, A, row0, HDIM, N);     // TDM -> LDS (or copy fallback)

    const int lane = threadIdx.x & 31;
    const int wv   = threadIdx.x >> 5;
    const int hi   = lane >> 4;
    const int m    = lane & 15;
    const int ncol = wv * 16 + m;              // 8 waves cover all 128 cols

    v8f acc;
    const float bv = bias[ncol];
#pragma unroll
    for (int i = 0; i < 8; ++i) acc[i] = bv;

    const bf16_t* arow = sA + m * HDIM + hi * 8;
    const bf16_t* brow = Bt + (size_t)ncol * HDIM + hi * 8;

#pragma unroll
    for (int kt = 0; kt < HDIM / 32; ++kt) {
        ABfrag a, b;
        a.h[0] = *(const v8bf*)(arow + kt * 32);
        a.h[1] = *(const v8bf*)(arow + kt * 32 + 16);
        b.h[0] = *(const v8bf*)(brow + kt * 32);
        b.h[1] = *(const v8bf*)(brow + kt * 32 + 16);
        if (kt + 1 < HDIM / 32)
            __builtin_prefetch(brow + (kt + 1) * 32, 0, 1);
        acc = __builtin_amdgcn_wmma_f32_16x16x32_bf16(
            false, a.v, false, b.v, (short)0, acc, false, false);
    }

#pragma unroll
    for (int v = 0; v < 8; ++v) {
        int r = row0 + v + hi * 8;
        if (r < N) out[(size_t)r * ODIM + ncol] = acc[v];
    }
}

// ---------------------------------------------------------------------------
// Launcher.  d_in: x, edge_index(int64), w1, b1, w2, b2.  d_out: f32 [N,128].
// Workspace layout (bytes): sum[N*C] | cnt[N] | hcat[N*256 bf16] |
//   w1t[512*256 bf16] | w2t[128*512 bf16] | h[N*512 bf16]   (~21 MB)
// ---------------------------------------------------------------------------
extern "C" void kernel_launch(void* const* d_in, const int* in_sizes, int n_in,
                              void* d_out, int out_size, void* d_ws, size_t ws_size,
                              hipStream_t stream) {
    (void)n_in; (void)out_size; (void)ws_size;

    const float*     x  = (const float*)d_in[0];
    const long long* ei = (const long long*)d_in[1];
    const float*     w1 = (const float*)d_in[2];
    const float*     b1 = (const float*)d_in[3];
    const float*     w2 = (const float*)d_in[4];
    const float*     b2 = (const float*)d_in[5];
    float*           out = (float*)d_out;

    const int N = in_sizes[0] / C;
    const int E = in_sizes[1] / 2;

    char* ws = (char*)d_ws;
    size_t off = 0;
    float*  sum  = (float*)(ws + off);  off += (size_t)N * C * sizeof(float);
    float*  cnt  = (float*)(ws + off);  off += ((size_t)N * sizeof(float) + 15) & ~(size_t)15;
    const size_t zero_bytes = off;      // sum + cnt need zeroing every call
    bf16_t* hcat = (bf16_t*)(ws + off); off += (size_t)N * TWO_C * sizeof(bf16_t);
    bf16_t* w1t  = (bf16_t*)(ws + off); off += (size_t)HDIM * TWO_C * sizeof(bf16_t);
    bf16_t* w2t  = (bf16_t*)(ws + off); off += (size_t)ODIM * HDIM * sizeof(bf16_t);
    bf16_t* hbuf = (bf16_t*)(ws + off); off += (size_t)N * HDIM * sizeof(bf16_t);

    hipMemsetAsync(ws, 0, zero_bytes, stream);   // graph-capture safe

    // 1) scatter-sum + degree count
    {
        long long threads = (long long)E * 32;
        int blocks = (int)((threads + 255) / 256);
        sage_scatter_kernel<<<blocks, 256, 0, stream>>>(x, ei, sum, cnt, E);
    }
    // 2) weight transpose+convert (tiny)
    {
        int t1 = TWO_C * HDIM;
        sage_tconv_kernel<<<(t1 + 255) / 256, 256, 0, stream>>>(w1, w1t, TWO_C, HDIM);
        int t2 = HDIM * ODIM;
        sage_tconv_kernel<<<(t2 + 255) / 256, 256, 0, stream>>>(w2, w2t, HDIM, ODIM);
    }
    // 3) concat + mean, convert to bf16
    {
        int t = N * TWO_C;
        sage_hcat_kernel<<<(t + 255) / 256, 256, 0, stream>>>(x, sum, cnt, hcat, N);
    }
    // 4) GEMM1 (WMMA bf16) + bias + relu
    {
        dim3 grid((N + 15) / 16, HDIM / 128);
        sage_gemm1_kernel<<<grid, 256, 0, stream>>>(hcat, w1t, b1, hbuf, N);
    }
    // 5) GEMM2 (WMMA bf16) + bias -> f32 out
    {
        dim3 grid((N + 15) / 16, 1);
        sage_gemm2_kernel<<<grid, 256, 0, stream>>>(hbuf, w2t, b2, out, N);
    }
}
